// SimplifiedCodec_90881507983967
// MI455X (gfx1250) — matrix operand order repaired and checked by reading, not compiled, not run on hardware
//
#include <hip/hip_runtime.h>

// ---------------- problem constants (from reference) ----------------
#define T_IN   98304
#define HOP    768
#define S_FR   128          // conv2 output frames per batch
#define NB     2
#define NTOT   256          // NB * S_FR
#define NCB    14
#define DIM    512
#define VOCAB  1024
#define CD     7168         // NCB*DIM
#define TOUT   97537        // (S_FR-1)*HOP + 5 - 2*2

typedef _Float16 h16;
typedef __attribute__((ext_vector_type(16))) _Float16 v16h;
typedef __attribute__((ext_vector_type(8)))  _Float16 v8h;
typedef __attribute__((ext_vector_type(8)))  float    v8f;
typedef __attribute__((ext_vector_type(4)))  unsigned int u32x4;
typedef __attribute__((ext_vector_type(8)))  int i32x8;
typedef __attribute__((ext_vector_type(4)))  int i32x4;

__device__ __forceinline__ h16  f2h(float f) { return (h16)f; }
__device__ __forceinline__ float h2f(h16 h)  { return (float)h; }

// Compiler-level fence for DMA-written LDS: passing the LDS pointer into the
// asm makes the object escape so the "memory" clobber applies to it; without
// this the optimizer proves smem is never stored and folds ds_loads to undef.
__device__ __forceinline__ void dma_lds_fence(void* lds_ptr) {
  asm volatile("" : "+v"(lds_ptr) :: "memory");
}

// ---- TDM availability (device pass on gfx1250 only; host parse falls back) --
#if defined(__gfx1250__) && __has_builtin(__builtin_amdgcn_tensor_load_to_lds)
  #define USE_TDM 1
#else
  #define USE_TDM 0
#endif
#if __has_include(<hip/amd_detail/amd_gfx1250_TDM.h>)
  #define TDM_6ARG 1
#else
  #define TDM_6ARG 0
#endif

#if USE_TDM
// Issue one 2-D TDM tile load: rows x row_elems (f16) from gptr (row pitch
// stride_elems) into LDS byte offset lds_off. D# per CDNA5 ISA ch.8.
__device__ __forceinline__ void tdm_load_tile(unsigned lds_off, const h16* gptr,
                                              unsigned rows, unsigned row_elems,
                                              unsigned stride_elems) {
  unsigned long long ga = (unsigned long long)(const void*)gptr;
  u32x4 g0 = {};
  g0.x = 1u;                                              // count=1 (valid D#)
  g0.y = lds_off;                                         // lds_addr (bytes)
  g0.z = (unsigned)ga;                                    // global_addr[31:0]
  g0.w = (unsigned)((ga >> 32) & 0x01ffffffu) | (2u << 30); // addr[56:32]|type=2
  i32x8 g1 = {};
  unsigned td0 = stride_elems;                            // tensor_dim0
  unsigned td1 = rows;                                    // tensor_dim1
  g1[0] = (int)0x00010000u;                               // data_size = 2 bytes
  g1[1] = (int)((td0 & 0xffffu) << 16);                   // tensor_dim0[15:0]
  g1[2] = (int)((td0 >> 16) | ((td1 & 0xffffu) << 16));   // dim0 hi | dim1 lo
  g1[3] = (int)((td1 >> 16) | (row_elems << 16));         // dim1 hi | tile_dim0
  g1[4] = (int)rows;                                      // tile_dim1
  g1[5] = (int)stride_elems;                              // dim0_stride[31:0]
  g1[6] = 0;                                              // dim0_stride hi
  g1[7] = 0;
  i32x4 z4 = {};
#if TDM_6ARG
  i32x8 z8 = {};
  __builtin_amdgcn_tensor_load_to_lds(g0, g1, z4, z4, z8, 0);
#else
  __builtin_amdgcn_tensor_load_to_lds(g0, g1, z4, z4, 0);
#endif
}
#endif

// ---- WMMA fragment layouts (CDNA5 16x16x32 f16, wave32) ----
__device__ __forceinline__ v16h frag_a(const h16* A, int lda, int m, int half, int k0) {
  const h16* p = A + (size_t)m * lda + k0 + 16 * half;
  v8h lo = *(const v8h*)p;
  v8h hi = *(const v8h*)(p + 8);
  v16h r;
#pragma unroll
  for (int i = 0; i < 8; ++i) { r[i] = lo[i]; r[i + 8] = hi[i]; }
  return r;
}
__device__ __forceinline__ v16h frag_b(const h16* Bt, int ldb, int n, int half, int k0) {
  const h16* p = Bt + (size_t)n * ldb + k0 + 16 * half;
  v8h lo = *(const v8h*)p;
  v8h hi = *(const v8h*)(p + 8);
  v16h r;
#pragma unroll
  for (int i = 0; i < 8; ++i) { r[i] = lo[i]; r[i + 8] = hi[i]; }
  return r;
}

// ------------------- LDS-staged, double-buffered f16 WMMA GEMM ---------------
// C(MxN,f32)[+bias][relu] and/or CT[n*ldct+m](f16) = A(MxK,row) @ Bt(NxK,row)^T
// block = 256 thr = 8 waves; tile 128(M) x 64(N) x 32(K); grid=(N/64, M/128, Z).
#define A_H (128 * 32)              // h16 per A tile
#define B_H (64 * 32)               // h16 per B tile
#define BUF_H (A_H + B_H)           // 6144 h16 = 12288 B per buffer
__global__ void __launch_bounds__(256) wmma_gemm(
    const h16* __restrict__ A, const h16* __restrict__ Bt,
    float* __restrict__ C, h16* __restrict__ CT,
    int K, int lda, int ldb, int ldc, int ldct,
    long long aStr, long long bStr, long long cStr, long long ctStr,
    const float* __restrict__ bias, int relu)
{
  __shared__ __align__(16) h16 smem[2 * BUF_H];
  const int z = blockIdx.z;
  A  += (size_t)z * aStr;
  Bt += (size_t)z * bStr;
  if (C)  C  += (size_t)z * cStr;
  if (CT) CT += (size_t)z * ctStr;

  const int tid  = threadIdx.x;
  const int lane = tid & 31;
  const int wv   = tid >> 5;
  const int half = lane >> 4;
  const int nlo  = lane & 15;
  const int mblk = blockIdx.y * 128;
  const int nblk = blockIdx.x * 64;
  const int nsteps = K >> 5;

  v8f acc[4];
#pragma unroll
  for (int t = 0; t < 4; ++t)
#pragma unroll
    for (int i = 0; i < 8; ++i) acc[t][i] = 0.0f;

#if USE_TDM
  if (wv == 0) {                     // one wave drives the Tensor Data Mover
    tdm_load_tile(0u,            A  + (size_t)mblk * lda, 128, 32, (unsigned)lda);
    tdm_load_tile(A_H * 2u,      Bt + (size_t)nblk * ldb,  64, 32, (unsigned)ldb);
    __builtin_amdgcn_s_wait_tensorcnt(0);
  }
  dma_lds_fence(smem);               // DMA-written LDS escapes to the fence
  __syncthreads();
  for (int i = 0; i < nsteps; ++i) {
    const int cur = i & 1;
    if (wv == 0 && i + 1 < nsteps) { // DMA next tile while we compute
      const unsigned boff = (unsigned)((1 - cur) * (BUF_H * 2));
      const int k0n = (i + 1) << 5;
      tdm_load_tile(boff,           A  + (size_t)mblk * lda + k0n, 128, 32, (unsigned)lda);
      tdm_load_tile(boff + A_H * 2, Bt + (size_t)nblk * ldb + k0n,  64, 32, (unsigned)ldb);
    }
    const h16* As = smem + cur * BUF_H;
    const h16* Bs = As + A_H;
    // issue ALL fragment loads first so the WMMAs need only partial ds waits
    v16h a = frag_a(As, 32, wv * 16 + nlo, half, 0);
    v16h b[4];
#pragma unroll
    for (int t = 0; t < 4; ++t) b[t] = frag_b(Bs, 32, t * 16 + nlo, half, 0);
#pragma unroll
    for (int t = 0; t < 4; ++t)
      acc[t] = __builtin_amdgcn_wmma_f32_16x16x32_f16(
          false, a, false, b[t], (short)0, acc[t], false, false);
    if (wv == 0 && i + 1 < nsteps) __builtin_amdgcn_s_wait_tensorcnt(0);
    dma_lds_fence(smem);
    __syncthreads();
  }
#else
  // cooperative double-buffered global->regs->LDS pipeline (no TDM)
  const int ar0 = tid >> 2,        as0 = (tid & 3) * 8;          // A chunks 0..255
  const int ar1 = (tid + 256) >> 2, as1 = ((tid + 256) & 3) * 8; // A chunks 256..511
  const int br  = tid >> 2,        bs  = (tid & 3) * 8;          // B chunks 0..255
  const h16* Ag = A  + (size_t)mblk * lda;
  const h16* Bg = Bt + (size_t)nblk * ldb;
  v8h ra0 = *(const v8h*)(Ag + (size_t)ar0 * lda + as0);
  v8h ra1 = *(const v8h*)(Ag + (size_t)ar1 * lda + as1);
  v8h rb0 = *(const v8h*)(Bg + (size_t)br  * ldb + bs);
  *(v8h*)(smem + ar0 * 32 + as0) = ra0;
  *(v8h*)(smem + ar1 * 32 + as1) = ra1;
  *(v8h*)(smem + A_H + br * 32 + bs) = rb0;
  __syncthreads();
  for (int i = 0; i < nsteps; ++i) {
    const int cur = i & 1;
    if (i + 1 < nsteps) {
      const int k0n = (i + 1) << 5;
      ra0 = *(const v8h*)(Ag + (size_t)ar0 * lda + k0n + as0);
      ra1 = *(const v8h*)(Ag + (size_t)ar1 * lda + k0n + as1);
      rb0 = *(const v8h*)(Bg + (size_t)br  * ldb + k0n + bs);
    }
    const h16* As = smem + cur * BUF_H;
    const h16* Bs = As + A_H;
    v16h a = frag_a(As, 32, wv * 16 + nlo, half, 0);
    v16h b[4];
#pragma unroll
    for (int t = 0; t < 4; ++t) b[t] = frag_b(Bs, 32, t * 16 + nlo, half, 0);
#pragma unroll
    for (int t = 0; t < 4; ++t)
      acc[t] = __builtin_amdgcn_wmma_f32_16x16x32_f16(
          false, a, false, b[t], (short)0, acc[t], false, false);
    if (i + 1 < nsteps) {
      h16* Ad = smem + (1 - cur) * BUF_H;
      *(v8h*)(Ad + ar0 * 32 + as0) = ra0;
      *(v8h*)(Ad + ar1 * 32 + as1) = ra1;
      *(v8h*)(Ad + A_H + br * 32 + bs) = rb0;
    }
    __syncthreads();
  }
#endif

#pragma unroll
  for (int t = 0; t < 4; ++t) {
    const int n = nblk + t * 16 + nlo;
#pragma unroll
    for (int r = 0; r < 8; ++r) {
      const int m = mblk + wv * 16 + 8 * half + r;  // ISA C layout: M = r+8*half
      float v = acc[t][r];
      if (bias) v += bias[m];
      if (relu) v = fmaxf(v, 0.0f);
      if (C)  C[(size_t)m * ldc + n] = v;
      if (CT) CT[(size_t)n * ldct + m] = f2h(v);
    }
  }
}

// ------------- encoder front: fused conv1+relu -> im2col for conv2 ----
__global__ void enc_front(const float* __restrict__ audio, const float* __restrict__ ew1,
                          const float* __restrict__ eb1, h16* __restrict__ X2t)
{
  int idx = blockIdx.x * 256 + threadIdx.x;          // 256 frames * 5 taps * 256 ch
  if (idx >= NTOT * 5 * 256) return;
  int c  = idx & 255;
  int j  = (idx >> 8) % 5;
  int bs = (idx >> 8) / 5;                           // b*128+s
  int b  = bs >> 7, s = bs & 127;
  int t  = HOP * s - 2 + j;                          // conv2 pad=2 (zero-pad h1)
  float v = 0.0f;
  if (t >= 0 && t < T_IN) {
    v = eb1[c];
#pragma unroll
    for (int i = 0; i < 7; ++i) {                    // conv1 k=7 pad=3
      int ta = t + i - 3;
      float x = (ta >= 0 && ta < T_IN) ? audio[(size_t)b * T_IN + ta] : 0.0f;
      v += x * ew1[c * 7 + i];
    }
    v = fmaxf(v, 0.0f);
  }
  X2t[(size_t)bs * 1280 + j * 256 + c] = f2h(v);
}

__global__ void pack_w2(const float* __restrict__ ew2, h16* __restrict__ W2m) {
  int idx = blockIdx.x * 256 + threadIdx.x;          // 512*1280
  if (idx >= 512 * 1280) return;
  int o = idx / 1280, k = idx % 1280, j = k >> 8, c = k & 255;
  W2m[idx] = f2h(ew2[(size_t)o * 1280 + c * 5 + j]);
}

__global__ void pack_x3(const h16* __restrict__ H2bT, h16* __restrict__ X3t) {
  int idx = blockIdx.x * 256 + threadIdx.x;          // NTOT*1536
  if (idx >= NTOT * 1536) return;
  int n = idx / 1536, k = idx % 1536, tt = k >> 9, c = k & 511;
  int b = n >> 7, s = n & 127, sp = s + tt - 1;
  X3t[idx] = (sp >= 0 && sp < S_FR) ? H2bT[(size_t)(b * 128 + sp) * 512 + c] : f2h(0.0f);
}

__global__ void pack_w3(const float* __restrict__ ew3, h16* __restrict__ W3m) {
  int idx = blockIdx.x * 256 + threadIdx.x;          // 7168*1536
  if (idx >= CD * 1536) return;
  int o = idx / 1536, k = idx % 1536, tt = k >> 9, c = k & 511;
  W3m[idx] = f2h(ew3[(size_t)o * 1536 + c * 3 + tt]);
}

// ---------------- VQ ----------------
__global__ void pack_cb(const float* __restrict__ cb, h16* __restrict__ CBh) {
  int idx = blockIdx.x * 256 + threadIdx.x;          // 14*1024*512 (same layout)
  if (idx >= NCB * VOCAB * DIM) return;
  CBh[idx] = f2h(cb[idx]);
}
__global__ void cb_norm(const float* __restrict__ cb, float* __restrict__ cn) {
  int row = blockIdx.x * 256 + threadIdx.x;          // 14*1024 rows
  if (row >= NCB * VOCAB) return;
  const float* p = cb + (size_t)row * DIM;
  float s = 0.0f;
  for (int d = 0; d < DIM; ++d) s += p[d] * p[d];
  cn[row] = s;
}
__global__ void vq_argmin(const float* __restrict__ G, const float* __restrict__ cn,
                          int* __restrict__ codes) {
  int idx = blockIdx.x * 256 + threadIdx.x;          // 14*256
  if (idx >= NCB * NTOT) return;
  int k = idx >> 8, n = idx & 255;
  float best = 3.4e38f; int bv = 0;
  for (int v = 0; v < VOCAB; ++v) {
    float sc = cn[k * VOCAB + v] - 2.0f * G[((size_t)k * VOCAB + v) * NTOT + n];
    if (sc < best) { best = sc; bv = v; }
  }
  codes[idx] = bv;
}
__global__ void gather_q(const float* __restrict__ cb, const int* __restrict__ codes,
                         h16* __restrict__ Qt) {
  int idx = blockIdx.x * 256 + threadIdx.x;          // NTOT*7168
  if (idx >= NTOT * CD) return;
  int n = idx / CD, kk = idx % CD, k = kk >> 9, d = kk & 511;
  int code = codes[k * NTOT + n];
  Qt[idx] = f2h(cb[((size_t)k * VOCAB + code) * DIM + d]);
}

// ---------------- decoder ----------------
__global__ void pack_wd1(const float* __restrict__ dw1, h16* __restrict__ Wd1m) {
  int idx = blockIdx.x * 256 + threadIdx.x;          // 2560*7168
  if (idx >= 2560 * CD) return;
  int row = idx / CD, i = idx % CD, r = row >> 9, c = row & 511;
  Wd1m[idx] = f2h(dw1[(size_t)i * 2560 + c * 5 + r]);   // dw1[i, c, r]
}
__global__ void pack_wd2(const float* __restrict__ dw2, h16* __restrict__ Wd2m) {
  int idx = blockIdx.x * 256 + threadIdx.x;          // 256*2560, K = r*512+c
  if (idx >= 256 * 2560) return;
  int o = idx / 2560, k = idx % 2560, r = k >> 9, c = k & 511;
  Wd2m[idx] = f2h(dw2[(size_t)o * 2560 + c * 5 + r]);
}
__global__ void dec_bg2(const float* __restrict__ dw2, const float* __restrict__ db1,
                        const float* __restrict__ db2, float* __restrict__ bg2) {
  int o = threadIdx.x;                                // 256 threads
  float acc = db2[o];
  for (int c = 0; c < 512; ++c) {
    float abg = fmaxf(db1[c], 0.0f);
    const float* wp = dw2 + (size_t)o * 2560 + c * 5;
#pragma unroll
    for (int r = 0; r < 5; ++r) acc += wp[r] * abg;
  }
  bg2[o] = fmaxf(acc, 0.0f);
}
__global__ void dec_outbg(const float* __restrict__ bg2, const float* __restrict__ dw3,
                          const float* __restrict__ db3, float* __restrict__ outbg) {
  if (threadIdx.x != 0) return;
  float acc = db3[0];
  for (int c = 0; c < 256; ++c)
#pragma unroll
    for (int i = 0; i < 7; ++i) acc += dw3[c * 7 + i] * bg2[c];
  outbg[0] = tanhf(acc);
}

// d2 conv only near frame centers: exact a2 at t = HOP*s-4 .. +11 per frame.
__global__ void __launch_bounds__(256) dec_d2near(
    const float* __restrict__ Y1, const float* __restrict__ db1,
    const h16* __restrict__ Wd2m, const float* __restrict__ db2,
    h16* __restrict__ A2ex)
{
  __shared__ __align__(16) h16 patch[20 * 512];
  const int bs = blockIdx.x;                          // b*128+s
  const int s  = bs & 127;
  for (int i = threadIdx.x; i < 20 * 512; i += 256) {
    int u = i >> 9, c = i & 511;
    int t = HOP * s - 6 + u;
    float v = 0.0f;
    if (t >= 0 && t < TOUT) {
      int d = u - 4;                                  // covered tap iff 0<=d<=4
      v = (d >= 0 && d <= 4)
            ? fmaxf(Y1[((size_t)d * 512 + c) * NTOT + bs] + db1[c], 0.0f)
            : fmaxf(db1[c], 0.0f);                    // constant background
    }
    patch[i] = f2h(v);
  }
  __syncthreads();

  const int lane = threadIdx.x & 31;
  const int wv   = threadIdx.x >> 5;
  const int half = lane >> 4;
  const int nlo  = lane & 15;
  v8f acc[2];
#pragma unroll
  for (int mt = 0; mt < 2; ++mt)
#pragma unroll
    for (int i = 0; i < 8; ++i) acc[mt][i] = 0.0f;

  for (int r = 0; r < 5; ++r) {
    for (int c0 = 0; c0 < 512; c0 += 32) {
      int k0 = r * 512 + c0;
      const h16* bp = &patch[(nlo + r) * 512 + c0 + 16 * half];
      v8h blo = *(const v8h*)bp;
      v8h bhi = *(const v8h*)(bp + 8);
      v16h bfrag;
#pragma unroll
      for (int i = 0; i < 8; ++i) { bfrag[i] = blo[i]; bfrag[i + 8] = bhi[i]; }
      v16h afrag[2];
#pragma unroll
      for (int mt = 0; mt < 2; ++mt)
        afrag[mt] = frag_a(Wd2m, 2560, (wv * 2 + mt) * 16 + nlo, half, k0);
#pragma unroll
      for (int mt = 0; mt < 2; ++mt)
        acc[mt] = __builtin_amdgcn_wmma_f32_16x16x32_f16(
            false, afrag[mt], false, bfrag, (short)0, acc[mt], false, false);
    }
  }
#pragma unroll
  for (int mt = 0; mt < 2; ++mt) {
    int mbase = (wv * 2 + mt) * 16 + 8 * half;
    v8h outv;
#pragma unroll
    for (int r = 0; r < 8; ++r)
      outv[r] = f2h(fmaxf(acc[mt][r] + db2[mbase + r], 0.0f));
    *(v8h*)(A2ex + ((size_t)bs * 16 + nlo) * 256 + mbase) = outv;
  }
}

// final conv d3 + tanh: constant except 15 samples per 768-frame
__global__ void dec_final(const h16* __restrict__ A2ex, const float* __restrict__ bg2,
                          const float* __restrict__ outbg, const float* __restrict__ dw3,
                          const float* __restrict__ db3, float* __restrict__ out)
{
  int idx = blockIdx.x * 256 + threadIdx.x;
  if (idx >= NB * TOUT) return;
  int b = idx / TOUT, t = idx % TOUT;
  int s = (t + 384) / HOP; if (s > S_FR - 1) s = S_FR - 1;
  int dt = t - HOP * s;
  if (dt < -7 || dt > 7) { out[idx] = outbg[0]; return; }
  float acc = db3[0];
  for (int i = 0; i < 7; ++i) {
    int t2 = t + i - 3;
    if (t2 < 0 || t2 >= TOUT) continue;                // sequence zero-padding
    int r2 = t2 - HOP * s + 4;
    if (r2 >= 0 && r2 < 16) {
      const h16* row = A2ex + ((size_t)(b * 128 + s) * 16 + r2) * 256;
      for (int c = 0; c < 256; ++c) acc += h2f(row[c]) * dw3[c * 7 + i];
    } else {
      for (int c = 0; c < 256; ++c) acc += bg2[c] * dw3[c * 7 + i];
    }
  }
  out[idx] = tanhf(acc);
}

// ------------------------------ host ------------------------------
extern "C" void kernel_launch(void* const* d_in, const int* in_sizes, int n_in,
                              void* d_out, int out_size, void* d_ws, size_t ws_size,
                              hipStream_t stream) {
  (void)in_sizes; (void)n_in; (void)out_size; (void)ws_size;
  const float* audio = (const float*)d_in[0];
  const float* cb    = (const float*)d_in[1];
  const float* ew1 = (const float*)d_in[2];  const float* eb1 = (const float*)d_in[3];
  const float* ew2 = (const float*)d_in[4];  const float* eb2 = (const float*)d_in[5];
  const float* ew3 = (const float*)d_in[6];  const float* eb3 = (const float*)d_in[7];
  const float* dw1 = (const float*)d_in[8];  const float* db1 = (const float*)d_in[9];
  const float* dw2 = (const float*)d_in[10]; const float* db2 = (const float*)d_in[11];
  const float* dw3 = (const float*)d_in[12]; const float* db3 = (const float*)d_in[13];
  float* out = (float*)d_out;

  char* p = (char*)d_ws;
  auto carve = [&](size_t bytes) -> void* {
    void* r = (void*)p; p += (bytes + 255) & ~(size_t)255; return r;
  };
  h16*   X2t  = (h16*)  carve((size_t)NTOT * 1280 * 2);
  h16*   W2m  = (h16*)  carve((size_t)512 * 1280 * 2);
  h16*   H2bT = (h16*)  carve((size_t)NTOT * 512 * 2);
  h16*   X3t  = (h16*)  carve((size_t)NTOT * 1536 * 2);
  h16*   W3m  = (h16*)  carve((size_t)CD * 1536 * 2);
  h16*   Et   = (h16*)  carve((size_t)NTOT * CD * 2);
  h16*   CBh  = (h16*)  carve((size_t)NCB * VOCAB * DIM * 2);
  float* cn   = (float*)carve((size_t)NCB * VOCAB * 4);
  float* G    = (float*)carve((size_t)NCB * VOCAB * NTOT * 4);
  int*   codes= (int*)  carve((size_t)NCB * NTOT * 4);
  h16*   Qt   = (h16*)  carve((size_t)NTOT * CD * 2);
  h16*   Wd1m = (h16*)  carve((size_t)2560 * CD * 2);
  float* Y1   = (float*)carve((size_t)2560 * NTOT * 4);
  h16*   Wd2m = (h16*)  carve((size_t)256 * 2560 * 2);
  h16*   A2ex = (h16*)  carve((size_t)NTOT * 16 * 256 * 2);
  float* bg2  = (float*)carve(256 * 4);
  float* obg  = (float*)carve(256);

  auto blk = [](long long n) { return (unsigned)((n + 255) / 256); };

  // encoder
  pack_w2  <<<blk(512LL * 1280), 256, 0, stream>>>(ew2, W2m);
  enc_front<<<blk((long long)NTOT * 5 * 256), 256, 0, stream>>>(audio, ew1, eb1, X2t);
  wmma_gemm<<<dim3(4, 4, 1), 256, 0, stream>>>(W2m, X2t, nullptr, H2bT,
      1280, 1280, 1280, 0, 512, 0, 0, 0, 0, eb2, 1);                 // M=512
  pack_x3  <<<blk((long long)NTOT * 1536), 256, 0, stream>>>(H2bT, X3t);
  pack_w3  <<<blk((long long)CD * 1536), 256, 0, stream>>>(ew3, W3m);
  wmma_gemm<<<dim3(4, 56, 1), 256, 0, stream>>>(W3m, X3t, nullptr, Et,
      1536, 1536, 1536, 0, CD, 0, 0, 0, 0, eb3, 0);                  // M=7168
  // VQ
  pack_cb  <<<blk((long long)NCB * VOCAB * DIM), 256, 0, stream>>>(cb, CBh);
  cb_norm  <<<blk((long long)NCB * VOCAB), 256, 0, stream>>>(cb, cn);
  wmma_gemm<<<dim3(4, 8, NCB), 256, 0, stream>>>(CBh, Et, G, nullptr,
      512, 512, CD, NTOT, 0,
      (long long)VOCAB * DIM, 512, (long long)VOCAB * NTOT, 0, nullptr, 0);
  vq_argmin<<<blk((long long)NCB * NTOT), 256, 0, stream>>>(G, cn, codes);
  gather_q <<<blk((long long)NTOT * CD), 256, 0, stream>>>(cb, codes, Qt);
  // decoder (transposed conv as GEMM; stride-768 sparsity exploited)
  pack_wd1 <<<blk(2560LL * CD), 256, 0, stream>>>(dw1, Wd1m);
  wmma_gemm<<<dim3(4, 20, 1), 256, 0, stream>>>(Wd1m, Qt, Y1, nullptr,
      CD, CD, CD, NTOT, 0, 0, 0, 0, 0, nullptr, 0);                  // M=2560
  pack_wd2 <<<blk(256LL * 2560), 256, 0, stream>>>(dw2, Wd2m);
  dec_bg2  <<<1, 256, 0, stream>>>(dw2, db1, db2, bg2);
  dec_outbg<<<1, 32, 0, stream>>>(bg2, dw3, db3, obg);
  dec_d2near<<<NTOT, 256, 0, stream>>>(Y1, db1, Wd2m, db2, A2ex);
  dec_final<<<blk((long long)NB * TOUT), 256, 0, stream>>>(A2ex, bg2, obg, dw3, db3, out);
}
